// GAT_Numbering_Corrector_V3_35897336660280
// MI455X (gfx1250) — matrix operand order repaired and checked by reading, not compiled, not run on hardware
//
#include <hip/hip_runtime.h>
#include <hip/hip_bf16.h>
#include <math.h>

typedef __bf16 bf16_t;
typedef __attribute__((ext_vector_type(16))) __bf16 v16bf;
typedef __attribute__((ext_vector_type(8)))  float v8f;
typedef __attribute__((ext_vector_type(4)))  unsigned int u32x4;
typedef __attribute__((ext_vector_type(4)))  float f32x4;

#define NN   10000      // nodes
#define MP   10016      // nodes padded to multiple of 32 (GEMM row tiles)
#define NEg  160000     // raw edges
#define ETg  170000     // edges + self loops
#define HCg  1024       // HID*HEADS
#define INC  1030       // input channels
#define KP1  1056       // INC padded to multiple of 32

union FragB { v16bf v; u32x4 q[2]; };

// ---------------------------------------------------------------------------
// GEMM: C[M,Nout] = A_bf16[Mp,KP] @ BT_bf16[Npad,KP]^T + bias
// One wave computes a 32x64 tile (8 WMMAs / k-step): A frags reused 4x,
// B frags 2x. KP is a template constant and the k-loop is fully unrolled:
// every fragment is a fresh SSA value (no runtime-indexed arrays -> no
// scratch/LDS demotion), letting the scheduler pipeline loads of step s+1
// over the WMMAs of step s, and folding B sub-tile strides into the 24-bit
// instruction offsets. A and B share contiguous k order (k = 16*hi + s).
// ---------------------------------------------------------------------------
template <int KP>
__global__ __launch_bounds__(256) void gemm_bf16(
    const bf16_t* __restrict__ A, const bf16_t* __restrict__ BT,
    const float* __restrict__ bias, float* __restrict__ C,
    int M, int Mp, int Npad, int Nout)
{
  const int wid  = (int)((blockIdx.x * blockDim.x + threadIdx.x) >> 5);
  const int lane = (int)(threadIdx.x & 31);
  const int mt = Mp >> 5;                 // 32-row tiles
  const int tiles = mt * (Npad >> 6);
  if (wid >= tiles) return;               // wave-uniform: EXEC stays all-ones
  const int tm = wid % mt;                // consecutive waves share B panel
  const int tn = wid / mt;
  const int lo = lane & 15;
  const int hi = lane >> 4;

  v8f acc[8];
  #pragma unroll
  for (int t = 0; t < 8; ++t)
    #pragma unroll
    for (int i = 0; i < 8; ++i) acc[t][i] = 0.0f;

  const bf16_t* arow = A + (size_t)(tm * 32 + lo) * KP;
  const bf16_t* bb   = BT + (size_t)(tn * 64 + lo) * KP;

  constexpr int STEPS = KP / 32;
  #pragma unroll
  for (int s = 0; s < STEPS; ++s) {
    const int k0 = s * 32;
    FragB a0, a1, b0, b1, b2, b3;
    // A 16x32 bf16: lane holds row (lane&15); two contiguous 16B chunks.
    a0.q[0] = *(const u32x4*)(arow + k0 + hi * 16);
    a0.q[1] = *(const u32x4*)(arow + k0 + hi * 16 + 8);
    a1.q[0] = *(const u32x4*)(arow + 16 * KP + k0 + hi * 16);
    a1.q[1] = *(const u32x4*)(arow + 16 * KP + k0 + hi * 16 + 8);
    // B 32x16 bf16: lane holds column; one contiguous 32B chunk (BT=[N,K]).
    b0.q[0] = *(const u32x4*)(bb + k0 + hi * 16);
    b0.q[1] = *(const u32x4*)(bb + k0 + hi * 16 + 8);
    b1.q[0] = *(const u32x4*)(bb + 16 * KP + k0 + hi * 16);
    b1.q[1] = *(const u32x4*)(bb + 16 * KP + k0 + hi * 16 + 8);
    b2.q[0] = *(const u32x4*)(bb + 32 * KP + k0 + hi * 16);
    b2.q[1] = *(const u32x4*)(bb + 32 * KP + k0 + hi * 16 + 8);
    b3.q[0] = *(const u32x4*)(bb + 48 * KP + k0 + hi * 16);
    b3.q[1] = *(const u32x4*)(bb + 48 * KP + k0 + hi * 16 + 8);

    acc[0] = __builtin_amdgcn_wmma_f32_16x16x32_bf16(false, a0.v, false, b0.v, (short)0, acc[0], false, false);
    acc[1] = __builtin_amdgcn_wmma_f32_16x16x32_bf16(false, a0.v, false, b1.v, (short)0, acc[1], false, false);
    acc[2] = __builtin_amdgcn_wmma_f32_16x16x32_bf16(false, a0.v, false, b2.v, (short)0, acc[2], false, false);
    acc[3] = __builtin_amdgcn_wmma_f32_16x16x32_bf16(false, a0.v, false, b3.v, (short)0, acc[3], false, false);
    acc[4] = __builtin_amdgcn_wmma_f32_16x16x32_bf16(false, a1.v, false, b0.v, (short)0, acc[4], false, false);
    acc[5] = __builtin_amdgcn_wmma_f32_16x16x32_bf16(false, a1.v, false, b1.v, (short)0, acc[5], false, false);
    acc[6] = __builtin_amdgcn_wmma_f32_16x16x32_bf16(false, a1.v, false, b2.v, (short)0, acc[6], false, false);
    acc[7] = __builtin_amdgcn_wmma_f32_16x16x32_bf16(false, a1.v, false, b3.v, (short)0, acc[7], false, false);
  }

  // D 16x16 f32: lane = N (lo), VGPR r = M row (hi*8 + r)
  #pragma unroll
  for (int mi = 0; mi < 2; ++mi) {
    const int mbase = tm * 32 + mi * 16 + hi * 8;
    #pragma unroll
    for (int t = 0; t < 4; ++t) {
      const int n = tn * 64 + t * 16 + lo;
      if (n < Nout) {
        const float bv = bias[n];
        #pragma unroll
        for (int r = 0; r < 8; ++r) {
          const int m = mbase + r;
          if (m < M) C[(size_t)m * Nout + n] = acc[mi * 4 + t][r] + bv;
        }
      }
    }
  }
}

// ---------------------------------------------------------------------------
// f32 -> bf16 convert with row + K padding (rows_out x Kp, zero-filled)
// ---------------------------------------------------------------------------
__global__ void pad_to_bf16(const float* __restrict__ in, bf16_t* __restrict__ out,
                            int rows_real, int rows_out, int K, int Kp)
{
  long long idx = (long long)blockIdx.x * blockDim.x + threadIdx.x;
  long long tot = (long long)rows_out * Kp;
  if (idx >= tot) return;
  int r = (int)(idx / Kp), k = (int)(idx % Kp);
  out[idx] = (r < rows_real && k < K) ? (bf16_t)in[(size_t)r * K + k] : (bf16_t)0.0f;
}

// w[K,N] f32  ->  wT[Nr,Kp] bf16 (transposed, zero padded)
__global__ void transpose_to_bf16(const float* __restrict__ w, bf16_t* __restrict__ wT,
                                  int K, int N, int Kp, int Nr)
{
  long long idx = (long long)blockIdx.x * blockDim.x + threadIdx.x;
  long long tot = (long long)Nr * Kp;
  if (idx >= tot) return;
  int n = (int)(idx / Kp), k = (int)(idx % Kp);
  wT[idx] = (k < K && n < N) ? (bf16_t)w[(size_t)k * N + n] : (bf16_t)0.0f;
}

__global__ void zero_i32(int* __restrict__ p, int n)
{
  int i = blockIdx.x * blockDim.x + threadIdx.x;
  if (i < n) p[i] = 0;
}

// ---------------------------------------------------------------------------
// CSR build: degree histogram -> block scan -> fill (edge ids per dst)
// ---------------------------------------------------------------------------
__global__ void deg_kernel(const long long* __restrict__ ei, int* __restrict__ deg)
{
  int e = blockIdx.x * blockDim.x + threadIdx.x;
  if (e >= ETg) return;
  int dst = (e < NEg) ? (int)ei[NEg + e] : (e - NEg);
  atomicAdd(&deg[dst], 1);
}

__global__ __launch_bounds__(1024) void scan_kernel(const int* __restrict__ deg,
                                                    int* __restrict__ rowstart, int n)
{
  __shared__ int lds[1024];
  __shared__ int carry;
  int tid = threadIdx.x;
  if (tid == 0) carry = 0;
  __syncthreads();
  for (int base = 0; base < n; base += 1024) {
    int i = base + tid;
    int v = (i < n) ? deg[i] : 0;
    lds[tid] = v;
    __syncthreads();
    for (int off = 1; off < 1024; off <<= 1) {
      int t = (tid >= off) ? lds[tid - off] : 0;
      __syncthreads();
      lds[tid] += t;
      __syncthreads();
    }
    if (i < n) rowstart[i] = carry + lds[tid] - v;   // exclusive
    __syncthreads();
    if (tid == 0) carry += lds[1023];
    __syncthreads();
  }
  if (tid == 0) rowstart[n] = carry;
}

__global__ void fill_kernel(const long long* __restrict__ ei,
                            const int* __restrict__ rowstart,
                            int* __restrict__ cursor, int* __restrict__ csr)
{
  int e = blockIdx.x * blockDim.x + threadIdx.x;
  if (e >= ETg) return;
  int dst = (e < NEg) ? (int)ei[NEg + e] : (e - NEg);
  int pos = rowstart[dst] + atomicAdd(&cursor[dst], 1);
  csr[pos] = e;
}

// ---------------------------------------------------------------------------
// Attention logits: one wave per edge.
// e[edge][h] = sum_c leakyrelu(xl[src][h,c] + xr[dst][h,c]) * att[h,c]
// ---------------------------------------------------------------------------
__global__ __launch_bounds__(256) void edge_e_kernel(
    const long long* __restrict__ ei, const float* __restrict__ xl,
    const float* __restrict__ xr, const float* __restrict__ att,
    float* __restrict__ e)
{
  int wid  = (int)((blockIdx.x * blockDim.x + threadIdx.x) >> 5);
  int lane = (int)(threadIdx.x & 31);
  if (wid >= ETg) return;
  int src, dst;
  if (wid < NEg) { src = (int)ei[wid]; dst = (int)ei[NEg + wid]; }
  else           { src = dst = wid - NEg; }
  const float* pl = xl + (size_t)src * HCg;
  const float* pr = xr + (size_t)dst * HCg;
  float s[4] = {0.f, 0.f, 0.f, 0.f};
  #pragma unroll
  for (int i = 0; i < 32; ++i) {
    int c = lane + 32 * i;                 // head = i>>3 (lane-independent)
    float m = pl[c] + pr[c];
    m = (m > 0.f) ? m : 0.2f * m;
    s[i >> 3] += m * att[c];
  }
  #pragma unroll
  for (int h = 0; h < 4; ++h) {
    float v = s[h];
    #pragma unroll
    for (int off = 16; off >= 1; off >>= 1) v += __shfl_xor(v, off, 32);
    s[h] = v;
  }
  if (lane == 0) {
    float* o = e + (size_t)wid * 4;
    o[0] = s[0]; o[1] = s[1]; o[2] = s[2]; o[3] = s[3];
  }
}

// ---------------------------------------------------------------------------
// Segment softmax + aggregation, one wave per destination node (atomic-free).
// out[d] = sum_e alpha_e * xl[src_e] + bias ; full 1024-ch row in registers.
// ---------------------------------------------------------------------------
__global__ __launch_bounds__(256) void agg_kernel(
    const long long* __restrict__ ei, const int* __restrict__ rowstart,
    const int* __restrict__ csr, const float* __restrict__ e,
    const float* __restrict__ xl, const float* __restrict__ bias,
    float* __restrict__ out)
{
  int d    = (int)((blockIdx.x * blockDim.x + threadIdx.x) >> 5);
  int lane = (int)(threadIdx.x & 31);
  if (d >= NN) return;
  int s = rowstart[d], t = rowstart[d + 1];

  // pass 1: per-head max
  float mx[4] = {-INFINITY, -INFINITY, -INFINITY, -INFINITY};
  for (int j = s + lane; j < t; j += 32) {
    const float* ev = e + (size_t)csr[j] * 4;
    mx[0] = fmaxf(mx[0], ev[0]); mx[1] = fmaxf(mx[1], ev[1]);
    mx[2] = fmaxf(mx[2], ev[2]); mx[3] = fmaxf(mx[3], ev[3]);
  }
  #pragma unroll
  for (int h = 0; h < 4; ++h) {
    float v = mx[h];
    #pragma unroll
    for (int off = 16; off >= 1; off >>= 1) v = fmaxf(v, __shfl_xor(v, off, 32));
    mx[h] = v;
  }
  // pass 2: per-head denominator
  float sm[4] = {0.f, 0.f, 0.f, 0.f};
  for (int j = s + lane; j < t; j += 32) {
    const float* ev = e + (size_t)csr[j] * 4;
    sm[0] += expf(ev[0] - mx[0]); sm[1] += expf(ev[1] - mx[1]);
    sm[2] += expf(ev[2] - mx[2]); sm[3] += expf(ev[3] - mx[3]);
  }
  float inv[4];
  #pragma unroll
  for (int h = 0; h < 4; ++h) {
    float v = sm[h];
    #pragma unroll
    for (int off = 16; off >= 1; off >>= 1) v += __shfl_xor(v, off, 32);
    inv[h] = 1.0f / (v + 1e-16f);
  }
  // pass 3: weighted aggregation; lane owns channels lane+32*i
  float acc[32];
  #pragma unroll
  for (int i = 0; i < 32; ++i) acc[i] = 0.f;
  for (int j = s; j < t; ++j) {
    int eid  = csr[j];
    int srcn = (eid < NEg) ? (int)ei[eid] : (eid - NEg);
    if (j + 1 < t) {                       // prefetch next gather row
      int eidn  = csr[j + 1];
      int srcnn = (eidn < NEg) ? (int)ei[eidn] : (eidn - NEg);
      __builtin_prefetch(xl + (size_t)srcnn * HCg + lane * 4, 0, 0);
    }
    const float* ev = e + (size_t)eid * 4;
    float a0 = expf(ev[0] - mx[0]) * inv[0];
    float a1 = expf(ev[1] - mx[1]) * inv[1];
    float a2 = expf(ev[2] - mx[2]) * inv[2];
    float a3 = expf(ev[3] - mx[3]) * inv[3];
    const float* px = xl + (size_t)srcn * HCg;
    #pragma unroll
    for (int i = 0; i < 32; ++i) {
      float al = (i < 8) ? a0 : (i < 16) ? a1 : (i < 24) ? a2 : a3;
      acc[i] = fmaf(px[lane + 32 * i], al, acc[i]);
    }
  }
  float* po = out + (size_t)d * HCg;
  #pragma unroll
  for (int i = 0; i < 32; ++i) {
    int c = lane + 32 * i;
    po[c] = acc[i] + bias[c];
  }
}

// ---------------------------------------------------------------------------
// Fused LayerNorm + ELU + bf16 quantize (one block of 256 per node row).
// Rows >= nvalid write zeros (GEMM row padding).
// ---------------------------------------------------------------------------
__global__ __launch_bounds__(256) void ln_elu_to_bf16(
    const float* __restrict__ in, const float* __restrict__ w,
    const float* __restrict__ b, bf16_t* __restrict__ out, int nvalid)
{
  __shared__ float ls[8], lss[8];
  const int row = blockIdx.x;
  const int tid = threadIdx.x;
  const int c = tid * 4;
  if (row >= nvalid) {
    union { bf16_t h[4]; unsigned long long u; } z;
    z.u = 0ull;
    *(unsigned long long*)(out + (size_t)row * HCg + c) = z.u;
    return;
  }
  const float* p = in + (size_t)row * HCg;
  f32x4 v = *(const f32x4*)(p + c);
  float s  = v.x + v.y + v.z + v.w;
  float ss = v.x * v.x + v.y * v.y + v.z * v.z + v.w * v.w;
  #pragma unroll
  for (int off = 16; off >= 1; off >>= 1) {
    s  += __shfl_xor(s, off, 32);
    ss += __shfl_xor(ss, off, 32);
  }
  if ((tid & 31) == 0) { ls[tid >> 5] = s; lss[tid >> 5] = ss; }
  __syncthreads();
  float tot = 0.f, tots = 0.f;
  #pragma unroll
  for (int i = 0; i < 8; ++i) { tot += ls[i]; tots += lss[i]; }
  const float mu   = tot * (1.0f / HCg);
  const float var  = tots * (1.0f / HCg) - mu * mu;   // biased, as jnp.var
  const float rstd = rsqrtf(var + 1e-5f);
  f32x4 wv = *(const f32x4*)(w + c);
  f32x4 bv = *(const f32x4*)(b + c);
  float y[4];
  y[0] = (v.x - mu) * rstd * wv.x + bv.x;
  y[1] = (v.y - mu) * rstd * wv.y + bv.y;
  y[2] = (v.z - mu) * rstd * wv.z + bv.z;
  y[3] = (v.w - mu) * rstd * wv.w + bv.w;
  union { bf16_t h[4]; unsigned long long u; } pk;
  #pragma unroll
  for (int i = 0; i < 4; ++i) {
    float z = y[i];
    z = (z > 0.f) ? z : expm1f(z);   // ELU
    pk.h[i] = (bf16_t)z;
  }
  *(unsigned long long*)(out + (size_t)row * HCg + c) = pk.u;
}

// ELU + bf16 with zero tail padding (i >= n_real -> 0)
__global__ void elu_to_bf16(const float* __restrict__ in, bf16_t* __restrict__ out,
                            int n_real, int n_total)
{
  int i = blockIdx.x * blockDim.x + threadIdx.x;
  if (i >= n_total) return;
  float x = 0.f;
  if (i < n_real) {
    x = in[i];
    x = (x > 0.f) ? x : expm1f(x);
  }
  out[i] = (bf16_t)x;
}

// ---------------------------------------------------------------------------
// Launcher
// ---------------------------------------------------------------------------
extern "C" void kernel_launch(void* const* d_in, const int* in_sizes, int n_in,
                              void* d_out, int out_size, void* d_ws, size_t ws_size,
                              hipStream_t stream)
{
  const float*     x       = (const float*)d_in[0];
  const long long* ei      = (const long long*)d_in[1];
  const float*     c1_wl   = (const float*)d_in[2];
  const float*     c1_bl   = (const float*)d_in[3];
  const float*     c1_wr   = (const float*)d_in[4];
  const float*     c1_br   = (const float*)d_in[5];
  const float*     c1_att  = (const float*)d_in[6];
  const float*     c1_bias = (const float*)d_in[7];
  const float*     ln1_w   = (const float*)d_in[8];
  const float*     ln1_b   = (const float*)d_in[9];
  const float*     c2_wl   = (const float*)d_in[10];
  const float*     c2_bl   = (const float*)d_in[11];
  const float*     c2_wr   = (const float*)d_in[12];
  const float*     c2_br   = (const float*)d_in[13];
  const float*     c2_att  = (const float*)d_in[14];
  const float*     c2_bias = (const float*)d_in[15];
  const float*     ln2_w   = (const float*)d_in[16];
  const float*     ln2_b   = (const float*)d_in[17];
  const float*     cls_w1  = (const float*)d_in[18];
  const float*     cls_b1  = (const float*)d_in[19];
  const float*     cls_w2  = (const float*)d_in[20];
  const float*     cls_b2  = (const float*)d_in[21];

  // ---- workspace carving (aligned 256B) ----
  char* p = (char*)d_ws;
  auto carve = [&](size_t bytes) -> char* {
    char* r = p; p += (bytes + 255) & ~(size_t)255; return r;
  };
  bf16_t* xb   = (bf16_t*)carve((size_t)MP * KP1 * 2);   // x bf16, reused as hb
  bf16_t* w1lT = (bf16_t*)carve((size_t)1024 * KP1 * 2);
  bf16_t* w1rT = (bf16_t*)carve((size_t)1024 * KP1 * 2);
  bf16_t* w2lT = (bf16_t*)carve((size_t)1024 * 1024 * 2);
  bf16_t* w2rT = (bf16_t*)carve((size_t)1024 * 1024 * 2);
  bf16_t* wc1T = (bf16_t*)carve((size_t)256 * 1024 * 2);
  bf16_t* wc2T = (bf16_t*)carve((size_t)64 * 256 * 2);
  float*  xl   = (float*)carve((size_t)NN * HCg * 4);    // reused as z1
  float*  xr   = (float*)carve((size_t)NN * HCg * 4);    // reused as z1b
  float*  agg  = (float*)carve((size_t)NN * HCg * 4);
  float*  ebuf = (float*)carve((size_t)ETg * 4 * 4);
  int*    deg      = (int*)carve((size_t)NN * 4);
  int*    cursor   = (int*)carve((size_t)NN * 4);
  int*    rowstart = (int*)carve((size_t)(NN + 1) * 4);
  int*    csr      = (int*)carve((size_t)ETg * 4);
  bf16_t* hb  = xb;               // conv output activations, bf16, [MP,1024]
  float*  z1  = xl;               // classifier hidden, f32, [NN,256]
  bf16_t* z1b = (bf16_t*)xr;      // classifier hidden, bf16, [MP,256]

  const int T = 256;
  auto grid1 = [&](long long n) { return (unsigned)((n + T - 1) / T); };

  // ---- convert / transpose / pad ----
  pad_to_bf16<<<grid1((long long)MP * KP1), T, 0, stream>>>(x, xb, NN, MP, INC, KP1);
  transpose_to_bf16<<<grid1((long long)1024 * KP1), T, 0, stream>>>(c1_wl, w1lT, INC, 1024, KP1, 1024);
  transpose_to_bf16<<<grid1((long long)1024 * KP1), T, 0, stream>>>(c1_wr, w1rT, INC, 1024, KP1, 1024);
  transpose_to_bf16<<<grid1((long long)1024 * 1024), T, 0, stream>>>(c2_wl, w2lT, 1024, 1024, 1024, 1024);
  transpose_to_bf16<<<grid1((long long)1024 * 1024), T, 0, stream>>>(c2_wr, w2rT, 1024, 1024, 1024, 1024);
  transpose_to_bf16<<<grid1((long long)256 * 1024), T, 0, stream>>>(cls_w1, wc1T, 1024, 256, 1024, 256);
  transpose_to_bf16<<<grid1((long long)64 * 256), T, 0, stream>>>(cls_w2, wc2T, 256, 49, 256, 64);

  // ---- CSR build (shared by both convs) ----
  zero_i32<<<grid1(NN), T, 0, stream>>>(deg, NN);
  zero_i32<<<grid1(NN), T, 0, stream>>>(cursor, NN);
  deg_kernel<<<grid1(ETg), T, 0, stream>>>(ei, deg);
  scan_kernel<<<1, 1024, 0, stream>>>(deg, rowstart, NN);
  fill_kernel<<<grid1(ETg), T, 0, stream>>>(ei, rowstart, cursor, csr);

  const int mt = MP / 32;                                        // 313
  const unsigned gemm_blk_1024 = (unsigned)((mt * 16 + 7) / 8);  // Npad=1024
  const unsigned gemm_blk_256  = (unsigned)((mt * 4 + 7) / 8);
  const unsigned gemm_blk_64   = (unsigned)((mt * 1 + 7) / 8);
  const unsigned edge_blk = (unsigned)((long long)ETg * 32 / T);  // 21250
  const unsigned node_blk = (unsigned)((long long)NN * 32 / T);   // 1250

  // ---- conv1 ----
  gemm_bf16<KP1><<<gemm_blk_1024, T, 0, stream>>>(xb, w1lT, c1_bl, xl, NN, MP, 1024, 1024);
  gemm_bf16<KP1><<<gemm_blk_1024, T, 0, stream>>>(xb, w1rT, c1_br, xr, NN, MP, 1024, 1024);
  edge_e_kernel<<<edge_blk, T, 0, stream>>>(ei, xl, xr, c1_att, ebuf);
  agg_kernel<<<node_blk, T, 0, stream>>>(ei, rowstart, csr, ebuf, xl, c1_bias, agg);
  ln_elu_to_bf16<<<MP, T, 0, stream>>>(agg, ln1_w, ln1_b, hb, NN);

  // ---- conv2 ----
  gemm_bf16<1024><<<gemm_blk_1024, T, 0, stream>>>(hb, w2lT, c2_bl, xl, NN, MP, 1024, 1024);
  gemm_bf16<1024><<<gemm_blk_1024, T, 0, stream>>>(hb, w2rT, c2_br, xr, NN, MP, 1024, 1024);
  edge_e_kernel<<<edge_blk, T, 0, stream>>>(ei, xl, xr, c2_att, ebuf);
  agg_kernel<<<node_blk, T, 0, stream>>>(ei, rowstart, csr, ebuf, xl, c2_bias, agg);
  ln_elu_to_bf16<<<MP, T, 0, stream>>>(agg, ln2_w, ln2_b, hb, NN);

  // ---- classifier ----
  gemm_bf16<1024><<<gemm_blk_256, T, 0, stream>>>(hb, wc1T, cls_b1, z1, NN, MP, 256, 256);
  elu_to_bf16<<<grid1((long long)MP * 256), T, 0, stream>>>(z1, z1b, NN * 256, MP * 256);
  gemm_bf16<256><<<gemm_blk_64, T, 0, stream>>>(z1b, wc2T, cls_b2, (float*)d_out, NN, MP, 64, 49);
}
